// GNN_85401129713862
// MI455X (gfx1250) — matrix operand order repaired and checked by reading, not compiled, not run on hardware
//
#include <hip/hip_runtime.h>
#include <hip/hip_bf16.h>

#define NNODES 50000
#define NEDGES 800000
#define HID 128
#define HEADS 4
#define CH 32
#define MT_TILES ((NNODES + 15) / 16)
#define MROWS 4                                   // m-tiles (of 16 rows) per block
#define MT_PAD  ((((MT_TILES) + (MROWS) - 1) / (MROWS)) * (MROWS))

typedef __attribute__((ext_vector_type(16))) _Float16 v16h;
typedef __attribute__((ext_vector_type(8)))  float    v8f;

// ---- optional CDNA5 async global->LDS path (guarded; falls back to manual copy) ----
#if defined(__has_builtin)
#  if __has_builtin(__builtin_amdgcn_global_load_async_to_lds_b128) && \
      __has_builtin(__builtin_amdgcn_s_wait_asynccnt)
#    define HAVE_ASYNC_LDS 1
#  endif
#endif
#ifndef HAVE_ASYNC_LDS
#  define HAVE_ASYNC_LDS 0
#endif

// builtin signature (from clang diagnostic): (v4i AS1*, v4i AS3*, imm offset, imm cpol)
typedef int v4i_ __attribute__((vector_size(16)));
typedef __attribute__((address_space(1))) v4i_ gv4i;   // global
typedef __attribute__((address_space(3))) v4i_ lv4i;   // LDS

// ---- monotonic float <-> uint mapping for atomicMax on f32 ----
__device__ __forceinline__ unsigned f2mono(float f) {
    unsigned u = __float_as_uint(f);
    return (u & 0x80000000u) ? ~u : (u | 0x80000000u);
}
__device__ __forceinline__ float mono2f(unsigned u) {
    return __uint_as_float((u & 0x80000000u) ? (u & 0x7FFFFFFFu) : ~u);
}

// ---- pack activations f32 [M x Din] -> WMMA A-layout f16 tiles ----
__global__ void pack_a_kernel(const float* __restrict__ src, _Float16* __restrict__ Apack,
                              int Din, int M, int KT) {
    int blk = blockIdx.x;
    int mt = blk / KT, kt = blk - mt * KT;
    int lane = threadIdx.x;
    int m = mt * 16 + (lane & 15);
    int kb = (lane >= 16) ? 8 : 0;
    v16h v;
#pragma unroll
    for (int e = 0; e < 16; ++e) {
        int j = e >> 1;
        int kj = (j < 4) ? (2 * j) : (16 + 2 * (j - 4));
        int k = kt * 32 + kj + kb + (e & 1);
        float f = (m < M) ? src[(size_t)m * Din + k] : 0.0f;
        v[e] = (_Float16)f;
    }
    *(v16h*)&Apack[((size_t)(mt * KT + kt) * 32 + lane) * 16] = v;
}

// ---- pack weights f32 [Din x Dout] -> WMMA B-layout f16 tiles ----
__global__ void pack_w_kernel(const float* __restrict__ W, _Float16* __restrict__ Wpack,
                              int Dout, int NT) {
    int blk = blockIdx.x;            // kt*NT + nt
    int kt = blk / NT, nt = blk - kt * NT;
    int lane = threadIdx.x;
    int n = nt * 16 + (lane & 15);
    int kg = kt * 32 + ((lane >= 16) ? 8 : 0);
    v16h v;
#pragma unroll
    for (int e = 0; e < 16; ++e)
        v[e] = (_Float16)W[(size_t)(kg + e + ((e >= 8) ? 8 : 0)) * Dout + n];
    *(v16h*)&Wpack[((size_t)blk * 32 + lane) * 16] = v;
}

// ---- WMMA GEMM: out[M x Dout] = A(f16) @ W(f16) + bias, opt ReLU ----
// block = (32, NT); each wave owns one 16-col tile and MROWS row-tiles (B reused 4x).
// A panel (MROWS x K) staged to LDS via async global->LDS when available.
__global__ void gemm_wmma_kernel(const _Float16* __restrict__ Apack,
                                 const _Float16* __restrict__ Wpack,
                                 const float* __restrict__ bias,
                                 float* __restrict__ out,
                                 int M, int KT, int NT, int ldOut, int act) {
    __shared__ alignas(32) _Float16 ldsA[MROWS * 4 * 32 * 16];   // 16 KB max (K<=128)
    int lane = threadIdx.x;
    int nt = threadIdx.y;
    int mt0 = blockIdx.x * MROWS;
    int tid = nt * 32 + lane;
    int nthreads = blockDim.y * 32;
    const _Float16* gsrc = Apack + (size_t)mt0 * KT * 512;
    int nvec = MROWS * KT * 64;          // 16-byte chunks to stage

#if HAVE_ASYNC_LDS
    {
        gv4i* gp = (gv4i*)(char*)gsrc;
        lv4i* lp = (lv4i*)(char*)ldsA;
        for (int i = tid; i < nvec; i += nthreads)
            __builtin_amdgcn_global_load_async_to_lds_b128(gp + i, lp + i, 0, 0);
        __builtin_amdgcn_s_wait_asynccnt(0);
    }
    __syncthreads();
#else
    {
        const unsigned* gu = (const unsigned*)gsrc;
        unsigned* lu = (unsigned*)ldsA;
        for (int i = tid; i < nvec * 4; i += nthreads) lu[i] = gu[i];
    }
    __syncthreads();
#endif

    v8f acc[MROWS] = {};
    const v16h* Alds = (const v16h*)ldsA;
    const v16h* Bp = (const v16h*)Wpack;
    for (int kt = 0; kt < KT; ++kt) {
        v16h b = Bp[(kt * NT + nt) * 32 + lane];
#pragma unroll
        for (int r = 0; r < MROWS; ++r) {
            v16h a = Alds[(r * KT + kt) * 32 + lane];
            acc[r] = __builtin_amdgcn_wmma_f32_16x16x32_f16(false, a, false, b,
                                                            (short)0, acc[r], false, false);
        }
    }

    int n = lane & 15;
    int col = nt * 16 + n;
    float bv = bias ? bias[col] : 0.0f;
    int mofs = (lane >= 16) ? 8 : 0;
#pragma unroll
    for (int r = 0; r < MROWS; ++r) {
#pragma unroll
        for (int rr = 0; rr < 8; ++rr) {
            int row = (mt0 + r) * 16 + mofs + rr;
            if (row < M) {
                float v = acc[r][rr] + bv;
                if (act) v = fmaxf(v, 0.0f);
                out[(size_t)row * ldOut + col] = v;
            }
        }
    }
}

// ---- edge pass 1: logits + per-(dst,head) running max via monotonic atomicMax ----
__global__ void edge_logits_kernel(const float* __restrict__ q, const float* __restrict__ k,
                                   const int* __restrict__ ei, float* __restrict__ logits,
                                   unsigned* __restrict__ mMono) {
    long total = (long)NEDGES * HEADS;
    long stride = (long)gridDim.x * blockDim.x;
    for (long id = (long)blockIdx.x * blockDim.x + threadIdx.x; id < total; id += stride) {
        long nid = id + stride;
        if (nid < total) __builtin_prefetch(ei + (int)(nid >> 2), 0, 1);
        int e = (int)(id >> 2), h = (int)(id & 3);
        int s = ei[e], d = ei[NEDGES + e];
        const float4* qa = (const float4*)(q + (size_t)d * HID + h * CH);
        const float4* ka = (const float4*)(k + (size_t)s * HID + h * CH);
        float acc = 0.f;
#pragma unroll
        for (int i = 0; i < 8; ++i) {
            float4 a = qa[i], b = ka[i];
            acc += a.x * b.x + a.y * b.y + a.z * b.z + a.w * b.w;
        }
        float lg = acc * 0.17677669529663687f;   // 1/sqrt(32)
        logits[id] = lg;
        atomicMax(&mMono[(size_t)d * HEADS + h], f2mono(lg));
    }
}

// ---- convert monotonic max back to f32 in place; empty/non-finite -> 0 ----
__global__ void m_fin_kernel(unsigned* mm) {
    int i = blockIdx.x * blockDim.x + threadIdx.x;
    if (i < NNODES * HEADS) {
        unsigned u = mm[i];
        float f = (u == 0u) ? 0.0f : mono2f(u);
        if (!(fabsf(f) <= 3.402823466e38f)) f = 0.0f;   // catches NaN and Inf
        ((float*)mm)[i] = f;
    }
}

// ---- edge pass 2: e = exp(logit - m); s += e; out_un += e * v[src] ----
__global__ void edge_accum_kernel(const float* __restrict__ logits, const float* __restrict__ m,
                                  const float* __restrict__ v, const int* __restrict__ ei,
                                  float* __restrict__ s, float* __restrict__ outun) {
    long total = (long)NEDGES * HEADS;
    long stride = (long)gridDim.x * blockDim.x;
    for (long id = (long)blockIdx.x * blockDim.x + threadIdx.x; id < total; id += stride) {
        long nid = id + stride;
        if (nid < total) __builtin_prefetch(logits + nid, 0, 1);
        int e = (int)(id >> 2), h = (int)(id & 3);
        int sr = ei[e], d = ei[NEDGES + e];
        float w = expf(logits[id] - m[(size_t)d * HEADS + h]);
        atomicAdd(&s[(size_t)d * HEADS + h], w);
        const float4* va = (const float4*)(v + (size_t)sr * HID + h * CH);
        float* ob = outun + (size_t)d * HID + h * CH;
#pragma unroll
        for (int i = 0; i < 8; ++i) {
            float4 vv = va[i];
            atomicAdd(ob + i * 4 + 0, w * vv.x);
            atomicAdd(ob + i * 4 + 1, w * vv.y);
            atomicAdd(ob + i * 4 + 2, w * vv.z);
            atomicAdd(ob + i * 4 + 3, w * vv.w);
        }
    }
}

// ---- per-node epilogue: normalize, beta-gate, LayerNorm, ReLU, residual ----
__global__ void node_fin_kernel(const float* __restrict__ outun, const float* __restrict__ s,
                                const float* __restrict__ skip, const float* __restrict__ Wbeta,
                                const float* __restrict__ lng, const float* __restrict__ lnb,
                                const float* __restrict__ res, float* __restrict__ hout) {
    int node = blockIdx.x * blockDim.y + threadIdx.y;
    if (node >= NNODES) return;
    int lane = threadIdx.x;
    int c0 = lane * 4;
    int h = lane >> 3;

    float4 ou = *(const float4*)(outun + (size_t)node * HID + c0);
    float inv = 1.0f / (s[(size_t)node * HEADS + h] + 1e-16f);
    float4 o = { ou.x * inv, ou.y * inv, ou.z * inv, ou.w * inv };
    float4 sk = *(const float4*)(skip + (size_t)node * HID + c0);

    float4 w1 = *(const float4*)(Wbeta + c0);
    float4 w2 = *(const float4*)(Wbeta + 128 + c0);
    float4 w3 = *(const float4*)(Wbeta + 256 + c0);
    float g = o.x * w1.x + o.y * w1.y + o.z * w1.z + o.w * w1.w
            + sk.x * w2.x + sk.y * w2.y + sk.z * w2.z + sk.w * w2.w
            + (o.x - sk.x) * w3.x + (o.y - sk.y) * w3.y
            + (o.z - sk.z) * w3.z + (o.w - sk.w) * w3.w;
    for (int off = 16; off; off >>= 1) g += __shfl_xor(g, off, 32);
    float beta = 1.0f / (1.0f + expf(-g));

    float4 hc = { beta * sk.x + (1.f - beta) * o.x,
                  beta * sk.y + (1.f - beta) * o.y,
                  beta * sk.z + (1.f - beta) * o.z,
                  beta * sk.w + (1.f - beta) * o.w };

    float lsum = hc.x + hc.y + hc.z + hc.w;
    for (int off = 16; off; off >>= 1) lsum += __shfl_xor(lsum, off, 32);
    float mu = lsum * (1.0f / HID);
    float dx = hc.x - mu, dy = hc.y - mu, dz = hc.z - mu, dw = hc.w - mu;
    float vsum = dx * dx + dy * dy + dz * dz + dw * dw;
    for (int off = 16; off; off >>= 1) vsum += __shfl_xor(vsum, off, 32);
    float rstd = rsqrtf(vsum * (1.0f / HID) + 1e-5f);

    float4 gg = *(const float4*)(lng + c0);
    float4 bb = *(const float4*)(lnb + c0);
    float4 y = { fmaxf(dx * rstd * gg.x + bb.x, 0.f),
                 fmaxf(dy * rstd * gg.y + bb.y, 0.f),
                 fmaxf(dz * rstd * gg.z + bb.z, 0.f),
                 fmaxf(dw * rstd * gg.w + bb.w, 0.f) };
    if (res) {
        float4 r = *(const float4*)(res + (size_t)node * HID + c0);
        y.x += r.x; y.y += r.y; y.z += r.z; y.w += r.w;
    }
    *(float4*)(hout + (size_t)node * HID + c0) = y;
}

// ---- final 256 -> 1 projection ----
__global__ void mlp1_kernel(const float* __restrict__ hidden, const float* __restrict__ W,
                            const float* __restrict__ b, float* __restrict__ out) {
    int node = blockIdx.x * blockDim.x + threadIdx.x;
    if (node >= NNODES) return;
    const float4* ha = (const float4*)(hidden + (size_t)node * 256);
    const float4* wa = (const float4*)W;
    float acc = b[0];
    for (int i = 0; i < 64; ++i) {
        float4 hv = ha[i], wv = wa[i];
        acc += hv.x * wv.x + hv.y * wv.y + hv.z * wv.z + hv.w * wv.w;
    }
    out[node] = acc;
}

extern "C" void kernel_launch(void* const* d_in, const int* in_sizes, int n_in,
                              void* d_out, int out_size, void* d_ws, size_t ws_size,
                              hipStream_t stream) {
    (void)in_sizes; (void)n_in; (void)out_size; (void)ws_size;
    // Input order: x, edge_index, conv0(Wq,bq,Wk,bk,Wv,bv,Wskip,bskip,Wbeta),
    // conv1(...), conv2(...), W_mlp0, b_mlp0, W_mlp1, b_mlp1, ln0_g..ln2_b
    const float* x  = (const float*)d_in[0];
    const int*   ei = (const int*)d_in[1];
    const float* W_mlp0 = (const float*)d_in[29];
    const float* b_mlp0 = (const float*)d_in[30];
    const float* W_mlp1 = (const float*)d_in[31];
    const float* b_mlp1 = (const float*)d_in[32];

    char* wp = (char*)d_ws;
    auto carve = [&](size_t bytes) -> char* {
        char* p = wp;
        wp += (bytes + 255) & ~(size_t)255;
        return p;
    };
    const size_t NV = NNODES;
    _Float16* Apack  = (_Float16*)carve((size_t)MT_PAD * 4 * 512 * sizeof(_Float16));
    _Float16* Wpack  = (_Float16*)carve((size_t)4 * 16 * 512 * sizeof(_Float16));
    float*    qb     = (float*)carve(NV * HID * 4);
    float*    kb     = (float*)carve(NV * HID * 4);
    float*    vb     = (float*)carve(NV * HID * 4);
    float*    skipb  = (float*)carve(NV * HID * 4);
    float*    logitsb= (float*)carve((size_t)NEDGES * HEADS * 4);
    unsigned* mMono  = (unsigned*)carve(NV * HEADS * 4);
    float*    sbuf   = (float*)carve(NV * HEADS * 4);
    float*    outun  = (float*)carve(NV * HID * 4);
    float*    hA     = (float*)carve(NV * HID * 4);
    float*    hB     = (float*)carve(NV * HID * 4);
    float*    hidden = (float*)carve(NV * 256 * 4);

    auto gemm = [&](const float* W, const float* bias, int Din, int Dout, float* out, int act) {
        int KT = Din / 32, NT = Dout / 16;
        pack_w_kernel<<<KT * NT, 32, 0, stream>>>(W, Wpack, Dout, NT);
        gemm_wmma_kernel<<<MT_PAD / MROWS, dim3(32, NT), 0, stream>>>(
            Apack, Wpack, bias, out, NNODES, KT, NT, Dout, act);
    };

    const float* hin = x;
    int Din = 64;
    float* houts[3] = { hA, hB, hA };
    for (int layer = 0; layer < 3; ++layer) {
        int pb = 2 + layer * 9;
        int KT = Din / 32;
        pack_a_kernel<<<MT_TILES * KT, 32, 0, stream>>>(hin, Apack, Din, NNODES, KT);
        gemm((const float*)d_in[pb + 0], (const float*)d_in[pb + 1], Din, HID, qb, 0);
        gemm((const float*)d_in[pb + 2], (const float*)d_in[pb + 3], Din, HID, kb, 0);
        gemm((const float*)d_in[pb + 4], (const float*)d_in[pb + 5], Din, HID, vb, 0);
        gemm((const float*)d_in[pb + 6], (const float*)d_in[pb + 7], Din, HID, skipb, 0);

        (void)hipMemsetAsync(mMono, 0, NV * HEADS * 4, stream);
        (void)hipMemsetAsync(sbuf, 0, NV * HEADS * 4, stream);
        (void)hipMemsetAsync(outun, 0, NV * HID * 4, stream);

        edge_logits_kernel<<<12500, 256, 0, stream>>>(qb, kb, ei, logitsb, mMono);
        m_fin_kernel<<<(NNODES * HEADS + 255) / 256, 256, 0, stream>>>(mMono);
        edge_accum_kernel<<<12500, 256, 0, stream>>>(logitsb, (const float*)mMono, vb, ei,
                                                     sbuf, outun);
        const float* res = (layer == 0) ? nullptr : hin;
        node_fin_kernel<<<(NNODES + 7) / 8, dim3(32, 8), 0, stream>>>(
            outun, sbuf, skipb, (const float*)d_in[pb + 8],
            (const float*)d_in[33 + layer * 2], (const float*)d_in[34 + layer * 2],
            res, houts[layer]);
        hin = houts[layer];
        Din = HID;
    }

    // MLP head: relu(h @ W0 + b0) @ W1 + b1
    pack_a_kernel<<<MT_TILES * 4, 32, 0, stream>>>(hin, Apack, HID, NNODES, 4);
    gemm(W_mlp0, b_mlp0, HID, 256, hidden, 1);
    mlp1_kernel<<<(NNODES + 255) / 256, 256, 0, stream>>>(hidden, W_mlp1, b_mlp1, (float*)d_out);
}
